// ConsistencyLoss_70738111365306
// MI455X (gfx1250) — matrix operand order repaired and checked by reading, not compiled, not run on hardware
//
#include <hip/hip_runtime.h>

typedef __attribute__((ext_vector_type(16))) _Float16 v16h;
typedef __attribute__((ext_vector_type(8)))  float    v8f;

#define B_   16
#define M_   32
#define N_   65536
#define W_   256
#define C_   27
#define CP_  32
#define GY_  8                       // blocks per batch (hist kernel)
#define WPB_ 8                       // waves per block
#define KPW_ (N_ / (GY_ * WPB_))     // 1024 K-elements per wave
#define PFD_ 256                     // prefetch distance: 8 chunks = 1KB = 8 lines

// ---------------------------------------------------------------- zero ws
__global__ void zero_ws_k(float* ws, int n) {
    int i = blockIdx.x * blockDim.x + threadIdx.x;
    if (i < n) ws[i] = 0.0f;
}

// ------------------------------------------------------- A-tile (16x32 f16)
__device__ __forceinline__ v16h load_a_tile(const float* __restrict__ row,
                                            int kb, int half) {
    const float* p0 = row + kb + half * 16;       // elements 0..7  (K = half*16 + e)
    const float* p1 = row + kb + half * 16 + 8;   // elements 8..15 (K = half*16 + 8 + e)
    float4 f0 = *(const float4*)(p0);
    float4 f1 = *(const float4*)(p0 + 4);
    float4 f2 = *(const float4*)(p1);
    float4 f3 = *(const float4*)(p1 + 4);
    v16h a;
    a[0]=(_Float16)f0.x;  a[1]=(_Float16)f0.y;  a[2]=(_Float16)f0.z;  a[3]=(_Float16)f0.w;
    a[4]=(_Float16)f1.x;  a[5]=(_Float16)f1.y;  a[6]=(_Float16)f1.z;  a[7]=(_Float16)f1.w;
    a[8]=(_Float16)f2.x;  a[9]=(_Float16)f2.y;  a[10]=(_Float16)f2.z; a[11]=(_Float16)f2.w;
    a[12]=(_Float16)f3.x; a[13]=(_Float16)f3.y; a[14]=(_Float16)f3.z; a[15]=(_Float16)f3.w;
    return a;
}

// -------------------------------------------- B-tiles (32x16 one-hot, f16)
__device__ __forceinline__ void make_b_tiles(const int* __restrict__ sp, int cid,
                                             v16h& b0, v16h& b1) {
    const int4* q = (const int4*)sp;              // 16 consecutive sem values
    int4 s0 = q[0], s1 = q[1], s2 = q[2], s3 = q[3];
    int sv[16] = { s0.x,s0.y,s0.z,s0.w, s1.x,s1.y,s1.z,s1.w,
                   s2.x,s2.y,s2.z,s2.w, s3.x,s3.y,s3.z,s3.w };
#pragma unroll
    for (int e = 0; e < 16; ++e) {
        b0[e] = (sv[e] == cid)        ? (_Float16)1.0f : (_Float16)0.0f;
        b1[e] = (sv[e] == cid + 16)   ? (_Float16)1.0f : (_Float16)0.0f;
    }
}

// ------------------------------------- class_hist via WMMA (B,M,CP) = A x onehot
__global__ __launch_bounds__(256)
void hist_wmma_k(const float* __restrict__ masks,
                 const int*   __restrict__ sem,
                 float*       __restrict__ hist) {
    __shared__ float lh[M_ * CP_];
    const int b   = blockIdx.x;
    const int tid = threadIdx.x;
    for (int i = tid; i < M_ * CP_; i += 256) lh[i] = 0.0f;
    __syncthreads();

    const int wave = tid >> 5;
    const int lane = tid & 31;
    const int half = lane >> 4;     // 0: lanes 0-15, 1: lanes 16-31
    const int mrow = lane & 15;     // A-matrix row id
    const int cid  = lane & 15;     // B/D column id
    const int k0   = (blockIdx.y * WPB_ + wave) * KPW_;

    const float* row0 = masks + (size_t)(b * M_ + mrow)      * N_;
    const float* row1 = masks + (size_t)(b * M_ + 16 + mrow) * N_;
    const int*   sb   = sem   + (size_t)b * N_;

    v8f acc00 = {}, acc01 = {}, acc10 = {}, acc11 = {};

    for (int kb = k0; kb < k0 + KPW_; kb += 32) {
        // Speculative streaming prefetch ~8 cachelines ahead; unconditional so
        // it lowers to a bare global_prefetch_b8 with an immediate offset (OOB
        // addresses past the row end are silently dropped per ISA 10.5).
        __builtin_prefetch(row0 + kb + PFD_, 0, 0);
        __builtin_prefetch(row1 + kb + PFD_, 0, 0);

        v16h a0 = load_a_tile(row0, kb, half);
        v16h a1 = load_a_tile(row1, kb, half);
        v16h bt0, bt1;
        make_b_tiles(sb + kb + half * 16, cid, bt0, bt1);

        acc00 = __builtin_amdgcn_wmma_f32_16x16x32_f16(false, a0, false, bt0, (short)0, acc00, false, false);
        acc01 = __builtin_amdgcn_wmma_f32_16x16x32_f16(false, a0, false, bt1, (short)0, acc01, false, false);
        acc10 = __builtin_amdgcn_wmma_f32_16x16x32_f16(false, a1, false, bt0, (short)0, acc10, false, false);
        acc11 = __builtin_amdgcn_wmma_f32_16x16x32_f16(false, a1, false, bt1, (short)0, acc11, false, false);
    }

    // C/D layout: VGPR r, lanes<16 -> M=r, lanes>=16 -> M=8+r; N = lane%16
#pragma unroll
    for (int r = 0; r < 8; ++r) {
        int mr = r + half * 8;
        atomicAdd(&lh[mr * CP_ + cid],              acc00[r]);
        atomicAdd(&lh[mr * CP_ + 16 + cid],         acc01[r]);
        atomicAdd(&lh[(16 + mr) * CP_ + cid],       acc10[r]);
        atomicAdd(&lh[(16 + mr) * CP_ + 16 + cid],  acc11[r]);
    }
    __syncthreads();
    for (int i = tid; i < M_ * CP_; i += 256)
        atomicAdd(&hist[(size_t)b * M_ * CP_ + i], lh[i]);
}

// ------------------------------- boundary IoU + depth-boundary coherence pass
__global__ __launch_bounds__(256)
void boundary_k(const int*   __restrict__ sem,
                const float* __restrict__ masks,
                const float* __restrict__ depth,
                float* __restrict__ bd,      // [B][2] = {inter, union}
                float* __restrict__ dbc_acc) {
    const int b = blockIdx.x, tid = threadIdx.x;
    const int*   s  = sem   + (size_t)b * N_;
    const float* m0 = masks + (size_t)b * M_ * N_;   // instance_masks[b,0,:]
    const float* dp = depth + (size_t)b * N_;

    float inter = 0.0f, uni = 0.0f, dsum = 0.0f;
    for (int i = tid; i < N_; i += 256) {
        int x = i & (W_ - 1), y = i >> 8;
        int sc = s[i];
        int sl = x ? s[i - 1]  : sc;
        int su = y ? s[i - W_] : sc;
        float semb = (sc != sl || sc != su) ? 1.0f : 0.0f;

        float mc  = m0[i];
        float mgx = x ? (mc - m0[i - 1])  : 0.0f;
        float mgy = y ? (mc - m0[i - W_]) : 0.0f;
        float instb = (fabsf(mgx) > 0.3f || fabsf(mgy) > 0.3f) ? 1.0f : 0.0f;

        inter += semb * instb;
        uni   += fmaxf(semb, instb);

        float dc  = dp[i];
        float dgx = x ? (dc - dp[i - 1])  : 0.0f;
        float dgy = y ? (dc - dp[i - W_]) : 0.0f;
        float s2  = dgx * dgx + dgy * dgy;
        float db  = sqrtf(fmaxf(s2, 1e-24f));
        float dbc = fminf(db, 2.0f);                 // clip(db, 0, 2), db >= 0
        dsum += (1.0f + 3.0f * dbc) * (1.0f - semb) * dbc;
    }

    __shared__ float ri[256], ru[256], rd[256];
    ri[tid] = inter; ru[tid] = uni; rd[tid] = dsum;
    __syncthreads();
    for (int st = 128; st > 0; st >>= 1) {
        if (tid < st) { ri[tid] += ri[tid + st]; ru[tid] += ru[tid + st]; rd[tid] += rd[tid + st]; }
        __syncthreads();
    }
    if (tid == 0) {
        bd[b * 2]     = ri[0];
        bd[b * 2 + 1] = ru[0];
        atomicAdd(dbc_acc, rd[0]);
    }
}

// -------------------------------------------------------------- finalize
__global__ __launch_bounds__(256)
void finalize_k(const float* __restrict__ hist,
                const float* __restrict__ bd,
                const float* __restrict__ dbc_acc,
                float* __restrict__ out) {
    const int tid = threadIdx.x;
    float ent = 0.0f;
    for (int p = tid; p < B_ * M_; p += 256) {
        const float* h = hist + (size_t)p * CP_;
        float msum = 1e-6f;
        for (int c = 0; c < C_; ++c) msum += h[c];
        float e = 0.0f;
        for (int c = 0; c < C_; ++c) {
            float pr = fminf(fmaxf(h[c] / msum, 1e-7f), 1.0f);
            if (pr > 1e-6f) e -= pr * logf(pr + 1e-10f);
        }
        ent += e;
    }
    __shared__ float r[256];
    r[tid] = ent;
    __syncthreads();
    for (int st = 128; st > 0; st >>= 1) {
        if (tid < st) r[tid] += r[tid + st];
        __syncthreads();
    }
    if (tid == 0) {
        float lu = r[0] / ((float)(B_ * M_) + 1e-8f);
        float iou = 0.0f;
        for (int b = 0; b < B_; ++b)
            iou += bd[b * 2] / (bd[b * 2 + 1] + 1e-8f);
        float lb = 1.0f - iou / (float)B_;
        float ld = dbc_acc[0] / (float)(B_ * N_);
        out[0] = lu;
        out[1] = lb;
        out[2] = ld;
        out[3] = 0.3f * lu + 0.2f * lb + 0.2f * ld;
    }
}

extern "C" void kernel_launch(void* const* d_in, const int* in_sizes, int n_in,
                              void* d_out, int out_size, void* d_ws, size_t ws_size,
                              hipStream_t stream) {
    const int*   sem   = (const int*)  d_in[0];   // (16, 65536) int32
    const float* masks = (const float*)d_in[1];   // (16, 32, 65536) f32
    const float* depth = (const float*)d_in[2];   // (16, 65536) f32
    float* out = (float*)d_out;                   // 4 scalars

    float* wsf  = (float*)d_ws;
    float* hist = wsf;                            // B*M*CP = 16384 floats
    float* bd   = wsf + B_ * M_ * CP_;            // 32 floats
    float* dbc  = bd + 2 * B_;                    // 1 float

    const int nz = B_ * M_ * CP_ + 2 * B_ + 1;
    zero_ws_k<<<(nz + 255) / 256, 256, 0, stream>>>(wsf, nz);
    hist_wmma_k<<<dim3(B_, GY_), 256, 0, stream>>>(masks, sem, hist);
    boundary_k<<<B_, 256, 0, stream>>>(sem, masks, depth, bd, dbc);
    finalize_k<<<1, 256, 0, stream>>>(hist, bd, dbc, out);
}
